// NonLocalBlock_33397665694171
// MI455X (gfx1250) — matrix operand order repaired and checked by reading, not compiled, not run on hardware
//
#include <hip/hip_runtime.h>

// =====================================================================
// NonLocalBlock (B=16, C=1024, H=W=32, hw==C) for MI455X / gfx1250.
//
// 5 GEMMs of 1024^3 per batch (~172 GFLOP total) + row softmax.
// All GEMMs run on v_wmma_f32_16x16x32_bf16 (bf16 operands, f32 acc).
// LDS tiles are stored in WMMA-fragment-ready order so every fragment
// is two aligned ds_load_b128 ops (no scalar gathers / v_mov_b16 packing).
// Scratch layout (needs 96 MB of d_ws):
//   [0,32MB)   phi   bf16  (B x 1024 x 1024)
//   [32,64MB)  theta bf16  -> reused as g^T after GEMM3
//   [64,96MB)  scores bf16 -> softmaxed in place to attn
// =====================================================================

typedef __attribute__((ext_vector_type(16))) __bf16 v16bf;
typedef __attribute__((ext_vector_type(8)))  float  v8f;

__device__ __forceinline__ __bf16 f2bf(float f) {
    union { float f; unsigned u; } x; x.f = f;
    unsigned r = x.u + 0x7fffu + ((x.u >> 16) & 1u);   // round-to-nearest-even
    unsigned short hv = (unsigned short)(r >> 16);
    __bf16 out; __builtin_memcpy(&out, &hv, 2); return out;
}
__device__ __forceinline__ float bf2f(__bf16 h) {
    unsigned short hv; __builtin_memcpy(&hv, &h, 2);
    union { unsigned u; float f; } x; x.u = ((unsigned)hv) << 16;
    return x.f;
}

constexpr int NDIM = 1024;             // all matrices are 1024 x 1024
constexpr int BM = 128, BN = 128, BK = 64;
constexpr int BKP = 80;                // padded K row: 160 B = 32-byte aligned rows

// A-operand intra-32 K permutation (ISA 7.12.2, 16-bit A 16x32):
// lane-half h reads layout slots [16h, 16h+16) ; element k (0..31) lands at:
//   grp = k>>3, pos = ((grp&1)<<4) | ((grp>>1)<<3) | (k&7)
__device__ __forceinline__ int a_perm(int k) {
    const int k32 = k & 31, grp = k32 >> 3;
    return (k & ~31) + (((grp & 1) << 4) | ((grp >> 1) << 3) | (k32 & 7));
}

// ---------------------------------------------------------------------
// Generic tiled bf16-WMMA GEMM:  O[m,n] = sum_k A[m,k] * B[k,n]
//   A_F32 / B_F32 : source dtype f32 (else bf16)
//   A_TRANS       : A element (m,k) lives at A[k*1024 + m]
//   STORE_TRANS   : write O (bf16) at [n*1024 + m]
//   OUT_F32       : O is f32 (else bf16)
//   ADD_RES       : O += res (f32), only with OUT_F32
// Block: 256 threads = 8 waves; macro tile 128x128; wave tile 64x32
// (4x2 WMMA 16x16 tiles); BK=64 -> 16 WMMAs between barriers.
// ---------------------------------------------------------------------
template<bool A_F32, bool B_F32, bool A_TRANS, bool STORE_TRANS,
         bool OUT_F32, bool ADD_RES>
__global__ __launch_bounds__(256)
void gemm_wmma(const void* __restrict__ Av, const void* __restrict__ Bv,
               void* __restrict__ Ov,
               size_t a_bstride, size_t b_bstride, size_t o_bstride,
               const float* __restrict__ res) {
    __shared__ __attribute__((aligned(32))) __bf16 As[BM][BKP]; // frag-ready rows
    __shared__ __attribute__((aligned(32))) __bf16 Bs[BN][BKP]; // transposed tile

    const int tid  = threadIdx.x;
    const int lane = tid & 31;
    const int wave = tid >> 5;
    const int h    = (lane >> 4) & 1;   // lane half (0: lanes 0-15, 1: 16-31)
    const int ln   = lane & 15;
    const int wr   = wave >> 2;         // 0..1 : 64-row slab
    const int wc   = wave & 3;          // 0..3 : 32-col slab

    const int m0 = blockIdx.y * BM;
    const int n0 = blockIdx.x * BN;
    const size_t ab = (size_t)blockIdx.z * a_bstride;
    const size_t bb = (size_t)blockIdx.z * b_bstride;
    const size_t ob = (size_t)blockIdx.z * o_bstride;

    const float*  Af = (const float*)Av;
    const __bf16* Ah = (const __bf16*)Av;
    const float*  Bf = (const float*)Bv;
    const __bf16* Bh = (const __bf16*)Bv;

    const v8f vzero = {0.f, 0.f, 0.f, 0.f, 0.f, 0.f, 0.f, 0.f};
    v8f acc[4][2];
#pragma unroll
    for (int mi = 0; mi < 4; ++mi)
#pragma unroll
        for (int ni = 0; ni < 2; ++ni) acc[mi][ni] = vzero;

    for (int k0 = 0; k0 < NDIM; k0 += BK) {
        // ---- stage A tile (BM x BK) into LDS, fragment-permuted ----
        for (int i = tid; i < BM * BK; i += 256) {
            const int m = i >> 6, k = i & (BK - 1);
            const size_t gi = A_TRANS ? (size_t)(k0 + k) * NDIM + (m0 + m)
                                      : (size_t)(m0 + m) * NDIM + (k0 + k);
            float v;
            if (A_F32) {
                v = Af[ab + gi];
                if (k0 + BK < NDIM)
                    __builtin_prefetch(&Af[ab + gi + (A_TRANS ? (size_t)BK * NDIM : BK)], 0, 1);
            } else {
                v = bf2f(Ah[ab + gi]);
                if (k0 + BK < NDIM)
                    __builtin_prefetch(&Ah[ab + gi + (A_TRANS ? (size_t)BK * NDIM : BK)], 0, 1);
            }
            As[m][a_perm(k)] = f2bf(v);
        }
        // ---- stage B tile (BK x BN) transposed into LDS (fragment-permuted K order) ----
        for (int i = tid; i < BK * BN; i += 256) {
            const int k = i >> 7, n = i & (BN - 1);
            const size_t gi = (size_t)(k0 + k) * NDIM + (n0 + n);
            float v;
            if (B_F32) {
                v = Bf[bb + gi];
                if (k0 + BK < NDIM)
                    __builtin_prefetch(&Bf[bb + gi + (size_t)BK * NDIM], 0, 1);
            } else {
                v = bf2f(Bh[bb + gi]);
                if (k0 + BK < NDIM)
                    __builtin_prefetch(&Bh[bb + gi + (size_t)BK * NDIM], 0, 1);
            }
            Bs[n][a_perm(k)] = f2bf(v);
        }
        __syncthreads();

        // ---- two K=32 substeps; fragments are contiguous 32B LDS spans ----
#pragma unroll
        for (int ks = 0; ks < 2; ++ks) {
            const int fofs = ks * 32 + (h << 4);
            v16bf afrag[4], bfrag[2];
#pragma unroll
            for (int mi = 0; mi < 4; ++mi)
                afrag[mi] = *(const v16bf*)&As[wr * 64 + mi * 16 + ln][fofs];
#pragma unroll
            for (int ni = 0; ni < 2; ++ni)
                bfrag[ni] = *(const v16bf*)&Bs[wc * 32 + ni * 16 + ln][fofs];
#pragma unroll
            for (int mi = 0; mi < 4; ++mi)
#pragma unroll
                for (int ni = 0; ni < 2; ++ni)
                    acc[mi][ni] = __builtin_amdgcn_wmma_f32_16x16x32_bf16(
                        false, afrag[mi], false, bfrag[ni],
                        (short)0, acc[mi][ni], false, false);
        }
        __syncthreads();
    }

    // ---- store: D 16x16 f32 layout: M = r + 8h, N = lane&15 ----
    float*  Of = (float*)Ov;
    __bf16* Oh = (__bf16*)Ov;
#pragma unroll
    for (int mi = 0; mi < 4; ++mi) {
#pragma unroll
        for (int ni = 0; ni < 2; ++ni) {
            const int row0 = m0 + wr * 64 + mi * 16;
            const int col  = n0 + wc * 32 + ni * 16 + ln;
#pragma unroll
            for (int r = 0; r < 8; ++r) {
                const int m = row0 + r + (h << 3);
                float v = acc[mi][ni][r];
                if (OUT_F32) {
                    size_t oi = ob + (size_t)m * NDIM + col;
                    if (ADD_RES) v += res[oi];
                    Of[oi] = v;
                } else {
                    size_t oi = STORE_TRANS ? ob + (size_t)col * NDIM + m
                                            : ob + (size_t)m * NDIM + col;
                    Oh[oi] = f2bf(v);
                }
            }
        }
    }
}

// ---------------------------------------------------------------------
// In-place row softmax over bf16 rows of length 1024.
// One 256-thread block per row; block owns the row, so in-place is safe.
// ---------------------------------------------------------------------
__global__ __launch_bounds__(256)
void softmax_rows_bf16(void* sv) {
    __bf16* row = (__bf16*)sv + (size_t)blockIdx.x * NDIM;
    const int t = threadIdx.x;

    float v0 = bf2f(row[t]);
    float v1 = bf2f(row[t + 256]);
    float v2 = bf2f(row[t + 512]);
    float v3 = bf2f(row[t + 768]);

    __shared__ float red[256];
    red[t] = fmaxf(fmaxf(v0, v1), fmaxf(v2, v3));
    __syncthreads();
#pragma unroll
    for (int off = 128; off > 0; off >>= 1) {
        if (t < off) red[t] = fmaxf(red[t], red[t + off]);
        __syncthreads();
    }
    const float mx = red[0];
    __syncthreads();

    v0 = __expf(v0 - mx); v1 = __expf(v1 - mx);
    v2 = __expf(v2 - mx); v3 = __expf(v3 - mx);
    red[t] = v0 + v1 + v2 + v3;
    __syncthreads();
#pragma unroll
    for (int off = 128; off > 0; off >>= 1) {
        if (t < off) red[t] += red[t + off];
        __syncthreads();
    }
    const float inv = 1.0f / red[0];

    row[t]       = f2bf(v0 * inv);
    row[t + 256] = f2bf(v1 * inv);
    row[t + 512] = f2bf(v2 * inv);
    row[t + 768] = f2bf(v3 * inv);
}

// ---------------------------------------------------------------------
extern "C" void kernel_launch(void* const* d_in, const int* in_sizes, int n_in,
                              void* d_out, int out_size, void* d_ws, size_t ws_size,
                              hipStream_t stream) {
    const float* x       = (const float*)d_in[0];   // (16, 1024, 32, 32)
    const float* w_phi   = (const float*)d_in[1];   // (1024, 1024)
    const float* w_theta = (const float*)d_in[2];
    const float* w_mask  = (const float*)d_in[3];

    const size_t NSZ   = (size_t)NDIM * NDIM;       // elements per batch matrix
    const size_t BATCH = 16;
    const size_t bytes_per_buf = BATCH * NSZ * 2;   // 32 MB bf16 each

    if (ws_size < 3 * bytes_per_buf) return;        // needs 96 MB scratch

    char* ws      = (char*)d_ws;
    void* phi     = (void*)(ws);                       // bf16
    void* theta   = (void*)(ws + bytes_per_buf);       // bf16, reused as g^T
    void* scores  = (void*)(ws + 2 * bytes_per_buf);   // bf16, softmaxed in place

    dim3 grid(NDIM / BN, NDIM / BM, (unsigned)BATCH);  // (8, 8, 16)
    dim3 block(256);

    // 1) phi = W_phi * X   (f32 in, bf16 out)
    gemm_wmma<true, true, false, false, false, false>
        <<<grid, block, 0, stream>>>(w_phi, x, phi, 0, NSZ, NSZ, nullptr);
    // 2) theta = W_theta * X
    gemm_wmma<true, true, false, false, false, false>
        <<<grid, block, 0, stream>>>(w_theta, x, theta, 0, NSZ, NSZ, nullptr);
    // 3) S = theta^T * phi   (A transposed access)
    gemm_wmma<false, false, true, false, false, false>
        <<<grid, block, 0, stream>>>(theta, phi, scores, NSZ, NSZ, NSZ, nullptr);
    // 4) attn = softmax_rows(S), in place
    softmax_rows_bf16<<<dim3((unsigned)(BATCH * NDIM)), block, 0, stream>>>(scores);
    // 5) g^T = (attn * phi)^T  -> overwrite theta buffer
    gemm_wmma<false, false, false, true, false, false>
        <<<grid, block, 0, stream>>>(scores, phi, theta, NSZ, NSZ, NSZ, nullptr);
    // 6) out = W_mask * g^T + x   (f32 out with residual)
    gemm_wmma<true, false, false, false, true, true>
        <<<grid, block, 0, stream>>>(w_mask, theta, d_out, 0, NSZ, NSZ, x);
}
